// MF2Demo_26379689132515
// MI455X (gfx1250) — matrix operand order repaired and checked by reading, not compiled, not run on hardware
//
#include <hip/hip_runtime.h>
#include <hip/hip_bf16.h>

typedef __attribute__((ext_vector_type(16))) _Float16 v16h;
typedef __attribute__((ext_vector_type(8)))  _Float16 v8h;
typedef __attribute__((ext_vector_type(8)))  float    v8f;

#define WAVES 8          // 8 wave32 per block
#define TILE_M 128       // 8 waves * 16 rows
#define NBLK 64          // 8192 / 128
#define BROWS 8192
#define DIM 256

static __device__ __forceinline__ v8f wmma_f16(v16h a, v16h b, v8f c) {
    return __builtin_amdgcn_wmma_f32_16x16x32_f16(false, a, false, b, (short)0, c, false, false);
}

// ---------------------------------------------------------------------------
// Kernel 1: column sums of y [8192,18] -> ysum[18]  (deterministic tree)
// ---------------------------------------------------------------------------
__global__ void __launch_bounds__(256) mf2_ysum(const float* __restrict__ y,
                                                float* __restrict__ ysum) {
    __shared__ float part[256][18];
    float acc[18];
#pragma unroll
    for (int j = 0; j < 18; ++j) acc[j] = 0.f;
    for (int r = threadIdx.x; r < BROWS; r += 256) {
        const float* yr = y + r * 18;
#pragma unroll
        for (int j = 0; j < 18; ++j) acc[j] += yr[j];
    }
#pragma unroll
    for (int j = 0; j < 18; ++j) part[threadIdx.x][j] = acc[j];
    __syncthreads();
    if (threadIdx.x < 18) {
        float s = 0.f;
        for (int t = 0; t < 256; ++t) s += part[t][threadIdx.x];
        ysum[threadIdx.x] = s;
    }
}

// ---------------------------------------------------------------------------
// Kernel 2: fused MLP (WMMA f16/f32-acc) + logits over 256 cases + logsumexp
// ---------------------------------------------------------------------------
__global__ void __launch_bounds__(256) mf2_main(
    const float* __restrict__ x,
    const float* __restrict__ W1, const float* __restrict__ b1,
    const float* __restrict__ W2, const float* __restrict__ b2,
    const float* __restrict__ W3, const float* __restrict__ b3,
    const float* __restrict__ cases,
    float* __restrict__ Wuser_out,   // [8192,18]
    float* __restrict__ wsum_out,    // [NBLK,18] per-block colsum(W_user)
    float* __restrict__ lsum_out)    // [NBLK]    per-block sum(log_denom)
{
    // Weights transposed to Bt[col][K] f16 so B fragments are contiguous 32B.
    __shared__ __align__(32) _Float16 sW1t[64][256];      // 32 KB (cols 50..63 = 0)
    __shared__ __align__(32) _Float16 sW2t[32][64];       // 4 KB  (K 50..63 = 0)
    __shared__ __align__(32) _Float16 sW3t[32][32];       // 2 KB  (cols 18..31 = 0)
    __shared__ __align__(32) _Float16 sCas[256][32];      // 16 KB (K 18..31 = 0)
    __shared__ float sB1[64], sB2[32], sB3[32];
    __shared__ __align__(32) _Float16 sH1[WAVES][16][64]; // 16 KB  h1 (f16)
    __shared__ __align__(32) _Float16 sH2[WAVES][16][32]; // 8 KB   h2 then W_user (f16)
    __shared__ float sColSum[WAVES][18];
    __shared__ float sLog[WAVES];

    const int tid   = threadIdx.x;
    const int wave  = tid >> 5;
    const int lane  = tid & 31;
    const int half  = lane >> 4;     // half-wave id
    const int lan16 = lane & 15;
    const int kb    = half * 16;     // A-fragment K contiguous base
    const int blockRow = blockIdx.x * TILE_M;

    // ---- stage weights into LDS (f16, transposed, zero padded) ----
    for (int i = tid; i < 64 * 256; i += 256) {
        const int n = i >> 8, k = i & 255;
        sW1t[n][k] = (n < 50) ? (_Float16)W1[k * 50 + n] : (_Float16)0.f;
    }
    for (int i = tid; i < 32 * 64; i += 256) {
        const int n = i >> 6, k = i & 63;
        sW2t[n][k] = (k < 50) ? (_Float16)W2[k * 32 + n] : (_Float16)0.f;
    }
    for (int i = tid; i < 32 * 32; i += 256) {
        const int n = i >> 5, k = i & 31;
        sW3t[n][k] = (n < 18) ? (_Float16)W3[k * 18 + n] : (_Float16)0.f;
    }
    for (int i = tid; i < 256 * 32; i += 256) {
        const int n = i >> 5, k = i & 31;
        sCas[n][k] = (k < 18) ? (_Float16)cases[n * 18 + k] : (_Float16)0.f;
    }
    if (tid < 64) sB1[tid] = (tid < 50) ? b1[tid] : 0.f;
    if (tid < 32) sB2[tid] = b2[tid];
    if (tid < 32) sB3[tid] = (tid < 18) ? b3[tid] : 0.f;
    __syncthreads();

    // =========== GEMM1: [16x256] x [256x64]  (K = 8 steps of 32) ===========
    const int myrow = blockRow + wave * 16 + lan16;     // A row for this lane
    const float* xrow = x + (size_t)myrow * DIM;
    v8f acc1[4] = {};
#pragma unroll
    for (int k0 = 0; k0 < DIM; k0 += 32) {
        const float4 f0 = *(const float4*)&xrow[k0 + kb];
        const float4 f1 = *(const float4*)&xrow[k0 + kb + 4];
        const float4 f2 = *(const float4*)&xrow[k0 + kb + 8];
        const float4 f3 = *(const float4*)&xrow[k0 + kb + 12];
        v16h a;
        a[0]=(_Float16)f0.x; a[1]=(_Float16)f0.y; a[2]=(_Float16)f0.z; a[3]=(_Float16)f0.w;
        a[4]=(_Float16)f1.x; a[5]=(_Float16)f1.y; a[6]=(_Float16)f1.z; a[7]=(_Float16)f1.w;
        a[8]=(_Float16)f2.x; a[9]=(_Float16)f2.y; a[10]=(_Float16)f2.z; a[11]=(_Float16)f2.w;
        a[12]=(_Float16)f3.x; a[13]=(_Float16)f3.y; a[14]=(_Float16)f3.z; a[15]=(_Float16)f3.w;
#pragma unroll
        for (int nt = 0; nt < 4; ++nt) {
            const int col = nt * 16 + lan16;
            const v16h b = *(const v16h*)&sW1t[col][k0 + half * 16];
            acc1[nt] = wmma_f16(a, b, acc1[nt]);
        }
    }
    // sigmoid(h + b1) -> sH1 (f16)
#pragma unroll
    for (int nt = 0; nt < 4; ++nt) {
        const int col = nt * 16 + lan16;
        const float bb = sB1[col];
#pragma unroll
        for (int r = 0; r < 8; ++r) {
            float v = acc1[nt][r] + bb;
            v = 1.f / (1.f + __expf(-v));
            sH1[wave][r + 8 * half][col] = (_Float16)v;
        }
    }
    __syncthreads();

    // =========== GEMM2: [16x64] x [64x32]  (K = 2 steps of 32) ===========
    const _Float16* h1r = &sH1[wave][lan16][0];
    v8f acc2[2] = {};
#pragma unroll
    for (int k0 = 0; k0 < 64; k0 += 32) {
        const v8h lo = *(const v8h*)&h1r[k0 + kb];
        const v8h hi = *(const v8h*)&h1r[k0 + 8 + kb];
        const v16h a = __builtin_shufflevector(lo, hi, 0,1,2,3,4,5,6,7,8,9,10,11,12,13,14,15);
#pragma unroll
        for (int nt = 0; nt < 2; ++nt) {
            const int col = nt * 16 + lan16;
            const v16h b = *(const v16h*)&sW2t[col][k0 + half * 16];
            acc2[nt] = wmma_f16(a, b, acc2[nt]);
        }
    }
#pragma unroll
    for (int nt = 0; nt < 2; ++nt) {
        const int col = nt * 16 + lan16;
        const float bb = sB2[col];
#pragma unroll
        for (int r = 0; r < 8; ++r) {
            float v = acc2[nt][r] + bb;
            v = 1.f / (1.f + __expf(-v));
            sH2[wave][r + 8 * half][col] = (_Float16)v;
        }
    }
    __syncthreads();

    // =========== GEMM3: [16x32] x [32x32] -> W_user (cols 18..31 exactly 0) ===========
    const _Float16* h2r = &sH2[wave][lan16][0];
    v8f acc3[2] = {};
    {
        const v8h lo = *(const v8h*)&h2r[kb];
        const v8h hi = *(const v8h*)&h2r[8 + kb];
        const v16h a = __builtin_shufflevector(lo, hi, 0,1,2,3,4,5,6,7,8,9,10,11,12,13,14,15);
#pragma unroll
        for (int nt = 0; nt < 2; ++nt) {
            const int col = nt * 16 + lan16;
            const v16h b = *(const v16h*)&sW3t[col][half * 16];
            acc3[nt] = wmma_f16(a, b, acc3[nt]);
        }
    }
    __syncthreads();   // done reading sH2 as h2; about to overwrite with W_user f16

    // W_user = acc3 + b3; write to global out, stash f16 copy (reusing sH2),
    // and reduce per-block column sums for the loss term.
#pragma unroll
    for (int nt = 0; nt < 2; ++nt) {
        const int col = nt * 16 + lan16;
        const float bb = sB3[col];               // 0 for col >= 18
        float ps = 0.f;
#pragma unroll
        for (int r = 0; r < 8; ++r) {
            const int row = r + 8 * half;
            const float v = acc3[nt][r] + bb;    // exactly 0 for col >= 18
            sH2[wave][row][col] = (_Float16)v;
            if (col < 18) {
                Wuser_out[(blockRow + wave * 16 + row) * 18 + col] = v;
                ps += v;
            }
        }
        ps += __shfl_xor(ps, 16, 32);            // combine both halves (rows 0-7 + 8-15)
        if (col < 18 && half == 0) sColSum[wave][col] = ps;
    }
    __syncthreads();

    // =========== GEMM4: logits = W_user x cases^T  (16 N-tiles, K=32 zero-padded) ===========
    const _Float16* wur = &sH2[wave][lan16][0];
    const v8h wlo = *(const v8h*)&wur[kb];
    const v8h whi = *(const v8h*)&wur[8 + kb];
    const v16h a4 = __builtin_shufflevector(wlo, whi, 0,1,2,3,4,5,6,7,8,9,10,11,12,13,14,15);

    float mx[8], sm[8];
#pragma unroll
    for (int r = 0; r < 8; ++r) { mx[r] = -3.0e38f; sm[r] = 0.f; }
#pragma unroll
    for (int nt = 0; nt < 16; ++nt) {
        const int col = nt * 16 + lan16;
        const v16h b = *(const v16h*)&sCas[col][half * 16];
        v8f c = {};
        c = wmma_f16(a4, b, c);
#pragma unroll
        for (int r = 0; r < 8; ++r) {            // online logsumexp per lane
            const float v = c[r];
            const float m2 = fmaxf(mx[r], v);
            sm[r] = sm[r] * __expf(mx[r] - m2) + __expf(v - m2);
            mx[r] = m2;
        }
    }
    // combine across the 16 lanes of each half-wave (columns of the tile row)
#pragma unroll
    for (int off = 1; off < 16; off <<= 1) {
#pragma unroll
        for (int r = 0; r < 8; ++r) {
            const float m2 = __shfl_xor(mx[r], off, 32);
            const float s2 = __shfl_xor(sm[r], off, 32);
            const float mm = fmaxf(mx[r], m2);
            sm[r] = sm[r] * __expf(mx[r] - mm) + s2 * __expf(m2 - mm);
            mx[r] = mm;
        }
    }
    float lsum = 0.f;
#pragma unroll
    for (int r = 0; r < 8; ++r) lsum += mx[r] + __logf(sm[r]);  // 8 rows of this half
    lsum += __shfl_xor(lsum, 16, 32);                           // + other half's 8 rows
    if (lane == 0) sLog[wave] = lsum;
    __syncthreads();

    if (tid < 18) {
        float s = 0.f;
#pragma unroll
        for (int w = 0; w < WAVES; ++w) s += sColSum[w][tid];
        wsum_out[blockIdx.x * 18 + tid] = s;
    }
    if (tid == 255) {
        float s = 0.f;
#pragma unroll
        for (int w = 0; w < WAVES; ++w) s += sLog[w];
        lsum_out[blockIdx.x] = s;
    }
}

// ---------------------------------------------------------------------------
// Kernel 3: loss = sum(log_denom) - dot(colsum(W_user), colsum(y))
// ---------------------------------------------------------------------------
__global__ void mf2_final(const float* __restrict__ ysum,
                          const float* __restrict__ wsum,
                          const float* __restrict__ lsum,
                          float* __restrict__ loss_out) {
    if (threadIdx.x == 0 && blockIdx.x == 0) {
        float ws[18];
#pragma unroll
        for (int j = 0; j < 18; ++j) ws[j] = 0.f;
        for (int b = 0; b < NBLK; ++b)
            for (int j = 0; j < 18; ++j) ws[j] += wsum[b * 18 + j];
        float ls = 0.f;
        for (int b = 0; b < NBLK; ++b) ls += lsum[b];
        float dot = 0.f;
#pragma unroll
        for (int j = 0; j < 18; ++j) dot += ws[j] * ysum[j];
        *loss_out = ls - dot;     // -(S - log_denom) summed
    }
}

extern "C" void kernel_launch(void* const* d_in, const int* in_sizes, int n_in,
                              void* d_out, int out_size, void* d_ws, size_t ws_size,
                              hipStream_t stream) {
    const float* x     = (const float*)d_in[0];
    const float* y     = (const float*)d_in[1];
    const float* W1    = (const float*)d_in[2];
    const float* b1    = (const float*)d_in[3];
    const float* W2    = (const float*)d_in[4];
    const float* b2    = (const float*)d_in[5];
    const float* W3    = (const float*)d_in[6];
    const float* b3    = (const float*)d_in[7];
    const float* cases = (const float*)d_in[8];

    float* out  = (float*)d_out;              // [8192*18] W_user, then loss scalar
    float* ws   = (float*)d_ws;
    float* ysum = ws;                         // 18
    float* wsum = ws + 32;                    // 64*18
    float* lsum = ws + 32 + NBLK * 18;        // 64

    mf2_ysum<<<1, 256, 0, stream>>>(y, ysum);
    mf2_main<<<NBLK, 256, 0, stream>>>(x, W1, b1, W2, b2, W3, b3, cases,
                                       out, wsum, lsum);
    mf2_final<<<1, 32, 0, stream>>>(ysum, wsum, lsum, out + BROWS * 18);
}
